// DCN_69063074119926
// MI455X (gfx1250) — compile-verified, matrix operand order checked by reading, not run on hardware
//
#include <hip/hip_runtime.h>
#include <hip/hip_bf16.h>

typedef __attribute__((ext_vector_type(4)))  _Float16 v4h;
typedef __attribute__((ext_vector_type(8)))  _Float16 v8h;
typedef __attribute__((ext_vector_type(16))) _Float16 v16h;
typedef __attribute__((ext_vector_type(8)))  float    v8f;

#define Bn 8
#define Cc 64
#define Oc 64
#define Hc 128
#define Wc 128
#define HW (Hc*Wc)
#define KK 9
#define KDIM (KK*Cc)      // 576 reduction length, r = kk*64 + c
#define KSTEPS (KDIM/32)  // 18 k-steps of 32 (f16 WMMA)

// ---------------------------------------------------------------------------
// Kernel 1: NCHW -> NHWC transpose of x (LDS tiled, coalesced in and out).
// ---------------------------------------------------------------------------
__global__ void __launch_bounds__(256)
dcn_transpose_kernel(const float* __restrict__ x, float* __restrict__ xT) {
    __shared__ float tile[64][65];
    int blk = blockIdx.x;            // Bn * Hc * (Wc/64) = 8*128*2 = 2048
    int b   = blk >> 8;              // 256 blocks per batch
    int rem = blk & 255;
    int h   = rem >> 1;
    int w0  = (rem & 1) * 64;
    int t   = threadIdx.x;
    const float* xb = x + (size_t)b * Cc * HW + h * Wc + w0;
    #pragma unroll
    for (int i = 0; i < 16; ++i) {           // 4096 elems, read coalesced in w
        int idx = t + i * 256;
        int c = idx >> 6, w = idx & 63;
        tile[c][w] = xb[(size_t)c * HW + w];
    }
    __syncthreads();
    float* ob = xT + ((size_t)b * HW + h * Wc + w0) * Cc;
    #pragma unroll
    for (int i = 0; i < 16; ++i) {           // write coalesced in c
        int idx = t + i * 256;
        int w = idx >> 6, c = idx & 63;
        ob[(size_t)w * Cc + c] = tile[c][w];
    }
}

// ---------------------------------------------------------------------------
// Kernel 2: pack f32 weights into f16 GEMM layouts.
//   wA [64][576]  : wA[o][kk*64+c]  = w_conv[o][c][kk]
//   wOM[32][576]  : rows 0-17 w_offset, 18-26 w_mod, 27-31 zero
// ---------------------------------------------------------------------------
__global__ void __launch_bounds__(256)
dcn_pack_kernel(const float* __restrict__ w_offset, const float* __restrict__ w_mod,
                const float* __restrict__ w_conv,
                _Float16* __restrict__ wA, _Float16* __restrict__ wOM) {
    int e = blockIdx.x * 256 + threadIdx.x;   // 216 blocks -> exactly 96*576
    if (e < Oc * KDIM) {
        int o = e / KDIM, r = e - o * KDIM;
        int kk = r >> 6, c = r & 63;
        wA[e] = (_Float16)w_conv[((size_t)o * Cc + c) * KK + kk];
    } else {
        int e2 = e - Oc * KDIM;
        int o = e2 / KDIM, r = e2 - o * KDIM;
        int kk = r >> 6, c = r & 63;
        float v = 0.f;
        if (o < 18)      v = w_offset[((size_t)o * Cc + c) * KK + kk];
        else if (o < 27) v = w_mod[((size_t)(o - 18) * Cc + c) * KK + kk];
        wOM[e2] = (_Float16)v;
    }
}

// A/B fragment helpers (16-bit WMMA, wave32).
// A (16x32 MxK): lane = row M (mod 16); lane-half selects K runs {0..7,16..23} vs {8..15,24..31}.
// B (32x16 KxN): lane = col N (mod 16); same K-run split by lane-half.
__device__ __forceinline__ v16h frag_from(const _Float16* p) {
    v8h lo = *(const v8h*)p;          // 8 contiguous K values
    v8h hi = *(const v8h*)(p + 16);   // +16 K offset
    return __builtin_shufflevector(lo, hi, 0,1,2,3,4,5,6,7,8,9,10,11,12,13,14,15);
}

// ---------------------------------------------------------------------------
// Kernel 3: offset + modulation conv as implicit-im2col f16 WMMA GEMM.
//   D[32 x 64px] = wOM[32 x 576] * im2col(xT)[576 x 64px]
// Epilogue emits absolute sampling coords P[b][k][{y,x}][pix] and mask Mk.
// ---------------------------------------------------------------------------
__global__ void __launch_bounds__(256)
dcn_offmask_kernel(const float* __restrict__ xT, const _Float16* __restrict__ wOM,
                   const float* __restrict__ b_offset, const float* __restrict__ b_mod,
                   float* __restrict__ P, float* __restrict__ Mk) {
    __shared__ _Float16 Bs[64][32];           // [pixel][k-slice] f16, 4 KB
    int blk = blockIdx.x;                     // 8*128*2 = 2048
    int b   = blk >> 8;
    int rem = blk & 255;
    int h   = rem >> 1;
    int w0  = (rem & 1) * 64;
    int t    = threadIdx.x;
    int wave = t >> 5, lane = t & 31;
    int mt = wave >> 2, nt = wave & 3;        // 2 M-tiles x 4 N-tiles = 8 waves
    int lh = lane >> 4, lm = lane & 15;
    int sn = t >> 2, sq = t & 3;              // staging: pixel sn, channel-octet sq
    const float* xb = xT + (size_t)b * HW * Cc;
    v8f acc = {0.f,0.f,0.f,0.f,0.f,0.f,0.f,0.f};
    for (int iter = 0; iter < KSTEPS; ++iter) {
        int kk  = iter >> 1;
        int ch0 = (iter & 1) * 32;
        int dy = kk / 3 - 1, dx = kk - (kk / 3) * 3 - 1;
        int y = h + dy, x = w0 + sn + dx;
        float v[8] = {0.f,0.f,0.f,0.f,0.f,0.f,0.f,0.f};
        if ((unsigned)y < (unsigned)Hc && (unsigned)x < (unsigned)Wc) {
            const float* p = xb + ((size_t)(y * Wc + x)) * Cc + ch0 + sq * 8;
            float4 f0 = *(const float4*)p;
            float4 f1 = *(const float4*)(p + 4);
            v[0]=f0.x; v[1]=f0.y; v[2]=f0.z; v[3]=f0.w;
            v[4]=f1.x; v[5]=f1.y; v[6]=f1.z; v[7]=f1.w;
        }
        __syncthreads();                      // previous k-step's reads done
        v8h hv;
        #pragma unroll
        for (int i = 0; i < 8; ++i) hv[i] = (_Float16)v[i];
        *(v8h*)&Bs[sn][sq * 8] = hv;
        __syncthreads();
        v16h af = frag_from(wOM + (size_t)(mt * 16 + lm) * KDIM + iter * 32 + lh * 8);
        v16h bf = frag_from(&Bs[nt * 16 + lm][lh * 8]);
        acc = __builtin_amdgcn_wmma_f32_16x16x32_f16(false, af, false, bf,
                                                     (short)0, acc, false, false);
    }
    // Epilogue: C/D layout -> VGPR j holds M = j + 8*laneHalf, N = lane&15.
    int n   = nt * 16 + lm;
    int wpx = w0 + n;
    int pix = h * Wc + wpx;
    #pragma unroll
    for (int j = 0; j < 8; ++j) {
        int m = mt * 16 + lh * 8 + j;
        float val = acc[j];
        if (m < 18) {
            val += b_offset[m];
            int k = m >> 1, comp = m & 1;     // off.reshape(B,9,2,H,W)
            float base = comp ? (float)(wpx + (k - (k / 3) * 3) - 1)
                              : (float)(h + k / 3 - 1);
            P[(((size_t)b * KK + k) * 2 + comp) * HW + pix] = base + val;
        } else if (m < 27) {
            int k = m - 18;
            val += b_mod[k];
            Mk[((size_t)b * KK + k) * HW + pix] = 2.f / (1.f + __expf(-val));
        }
    }
}

// ---------------------------------------------------------------------------
// Kernel 4: main DCN GEMM.
//   out[64 x 32px] = wA[64 x 576] * sampled[576 x 32px]
// sampled built on the fly: per (kk,pixel) 4 precomputed corner weights
// (mask & validity folded in) blending contiguous NHWC channel runs.
// ---------------------------------------------------------------------------
__global__ void __launch_bounds__(256)
dcn_main_kernel(const float* __restrict__ xT, const _Float16* __restrict__ wA,
                const float* __restrict__ P, const float* __restrict__ Mk,
                const float* __restrict__ b_conv, float* __restrict__ out) {
    __shared__ float    wls[KK][32][4];       // corner weights * mask (0 if invalid)
    __shared__ int      ils[KK][32][4];       // clamped NHWC element offsets
    __shared__ _Float16 Bs[32][32];           // sampled k-slice, 2 KB
    int blk = blockIdx.x;                     // 8*128*4 = 4096
    int b   = blk >> 9;
    int rem = blk & 511;
    int h   = rem >> 2;
    int w0  = (rem & 3) * 32;
    int t   = threadIdx.x;
    // ---- per-(k,pixel) sampling geometry (288 items) ----
    for (int it = t; it < KK * 32; it += 256) {
        int kk = it >> 5, n = it & 31;
        int pix = h * Wc + (w0 + n);
        size_t bk = (size_t)b * KK + kk;
        float py = P[(bk * 2 + 0) * HW + pix];
        float px = P[(bk * 2 + 1) * HW + pix];
        float mv = Mk[bk * HW + pix];
        float y0f = floorf(py), x0f = floorf(px);
        float ty = py - y0f, tx = px - x0f;
        int y0 = (int)y0f, x0 = (int)x0f;
        #pragma unroll
        for (int cy = 0; cy < 2; ++cy)
            #pragma unroll
            for (int cx = 0; cx < 2; ++cx) {
                int yy = y0 + cy, xx = x0 + cx;
                bool valid = (yy >= 0) && (yy <= Hc - 1) && (xx >= 0) && (xx <= Wc - 1);
                float wy = cy ? ty : 1.f - ty;
                float wx = cx ? tx : 1.f - tx;
                int yc = yy < 0 ? 0 : (yy > Hc - 1 ? Hc - 1 : yy);
                int xc = xx < 0 ? 0 : (xx > Wc - 1 ? Wc - 1 : xx);
                wls[kk][n][cy * 2 + cx] = valid ? mv * wy * wx : 0.f;
                ils[kk][n][cy * 2 + cx] = (yc * Wc + xc) * Cc;
            }
    }
    __syncthreads();
    int wave = t >> 5, lane = t & 31;
    int mt = wave >> 1, nt = wave & 1;        // 4 M-tiles x 2 N-tiles = 8 waves
    int lh = lane >> 4, lm = lane & 15;
    int sn = t >> 3, sblk = t & 7;            // staging: pixel sn, channel-quad sblk
    const float* xb = xT + (size_t)b * HW * Cc;
    v8f acc = {0.f,0.f,0.f,0.f,0.f,0.f,0.f,0.f};
    for (int iter = 0; iter < KSTEPS; ++iter) {
        int kk = iter >> 1;
        int c  = (iter & 1) * 32 + sblk * 4;
        float r0 = 0.f, r1 = 0.f, r2 = 0.f, r3 = 0.f;
        #pragma unroll
        for (int corner = 0; corner < 4; ++corner) {
            float wgt = wls[kk][sn][corner];
            const float* p = xb + ils[kk][sn][corner] + c;
            float4 vv = *(const float4*)p;    // L2-resident gather, 16B
            r0 += wgt * vv.x; r1 += wgt * vv.y;
            r2 += wgt * vv.z; r3 += wgt * vv.w;
        }
        __syncthreads();                      // previous k-step's reads done
        v4h hv;
        hv[0] = (_Float16)r0; hv[1] = (_Float16)r1;
        hv[2] = (_Float16)r2; hv[3] = (_Float16)r3;
        *(v4h*)&Bs[sn][sblk * 4] = hv;
        __syncthreads();
        v16h af = frag_from(wA + (size_t)(mt * 16 + lm) * KDIM + iter * 32 + lh * 8);
        v16h bf = frag_from(&Bs[nt * 16 + lm][lh * 8]);
        acc = __builtin_amdgcn_wmma_f32_16x16x32_f16(false, af, false, bf,
                                                     (short)0, acc, false, false);
    }
    // ---- epilogue: NCHW store + bias ----
    int wpx = w0 + nt * 16 + lm;
    float* op = out + (size_t)b * Oc * HW + h * Wc + wpx;
    #pragma unroll
    for (int j = 0; j < 8; ++j) {
        int o = mt * 16 + lh * 8 + j;
        op[(size_t)o * HW] = acc[j] + b_conv[o];
    }
}

// ---------------------------------------------------------------------------
// Workspace layout (all offsets 256B aligned):
//   xT  : B*HW*C floats            = 33,554,432 B
//   P   : B*9*2*HW floats          =  9,437,184 B
//   Mk  : B*9*HW floats            =  4,718,592 B
//   wA  : 64*576 f16               =     73,728 B
//   wOM : 32*576 f16               =     36,864 B
// ---------------------------------------------------------------------------
extern "C" void kernel_launch(void* const* d_in, const int* in_sizes, int n_in,
                              void* d_out, int out_size, void* d_ws, size_t ws_size,
                              hipStream_t stream) {
    const float* x        = (const float*)d_in[0];
    const float* w_offset = (const float*)d_in[1];
    const float* b_offset = (const float*)d_in[2];
    const float* w_mod    = (const float*)d_in[3];
    const float* b_mod    = (const float*)d_in[4];
    const float* w_conv   = (const float*)d_in[5];
    const float* b_conv   = (const float*)d_in[6];
    float* out = (float*)d_out;

    char* ws = (char*)d_ws;
    size_t off = 0;
    float* xT = (float*)(ws + off);            off += (size_t)Bn * HW * Cc * 4;
    float* P  = (float*)(ws + off);            off += (size_t)Bn * KK * 2 * HW * 4;
    float* Mk = (float*)(ws + off);            off += (size_t)Bn * KK * HW * 4;
    _Float16* wA  = (_Float16*)(ws + off);     off += (size_t)Oc * KDIM * 2;
    _Float16* wOM = (_Float16*)(ws + off);

    dcn_transpose_kernel<<<Bn * Hc * (Wc / 64), 256, 0, stream>>>(x, xT);
    dcn_pack_kernel<<<((Oc + 32) * KDIM) / 256, 256, 0, stream>>>(w_offset, w_mod, w_conv, wA, wOM);
    dcn_offmask_kernel<<<Bn * Hc * (Wc / 64), 256, 0, stream>>>(xT, wOM, b_offset, b_mod, P, Mk);
    dcn_main_kernel<<<Bn * Hc * (Wc / 32), 256, 0, stream>>>(xT, wA, P, Mk, b_conv, out);
}
